// AnomalyEncoder_69724499083284
// MI455X (gfx1250) — matrix operand order repaired and checked by reading, not compiled, not run on hardware
//
#include <hip/hip_runtime.h>

#define B_SZ 32
#define T_SZ 8192
#define OUTC 64
#define TPAD (T_SZ + 4)
#define KTOT 320   // 5 taps * 64 channels

typedef __attribute__((ext_vector_type(16))) __bf16 v16bf;
typedef __attribute__((ext_vector_type(8)))  __bf16 v8bf;
typedef __attribute__((ext_vector_type(8)))  float  v8f;

__device__ __forceinline__ unsigned short f32_to_bf16(float f) {
  union { float f; unsigned int u; } v; v.f = f;
  unsigned int u = v.u;
  u += 0x7FFFu + ((u >> 16) & 1u);   // round-to-nearest-even
  return (unsigned short)(u >> 16);
}

__device__ __forceinline__ float gridp(int j) { return 0.4f * (float)j - 2.2f; }

// Evaluate one KAN-MoE expert mixture for scalar x -> 32 outputs.
__device__ __forceinline__ void kan_eval(float x, const float* __restrict__ gate,
                                         const float* __restrict__ base,
                                         const float* __restrict__ spline,
                                         float* __restrict__ out) {
  // silu(x)
  float sil = x / (1.0f + __expf(-x));
  // Cox-de Boor B-splines, order 3, uniform grid g[j]=0.4j-2.2, 12 knots
  float bs[11];
#pragma unroll
  for (int j = 0; j < 11; ++j)
    bs[j] = (x >= gridp(j) && x < gridp(j + 1)) ? 1.0f : 0.0f;
#pragma unroll
  for (int ord = 1; ord <= 3; ++ord) {
    float inv = 1.0f / (0.4f * (float)ord);
#pragma unroll
    for (int i = 0; i < 11 - 1; ++i) {
      if (i < 11 - ord) {
        float left  = (x - gridp(i)) * inv;
        float right = (gridp(i + ord + 1) - x) * inv;
        bs[i] = left * bs[i] + right * bs[i + 1];
      }
    }
  }
  // softmax gate over 8 experts (logit = x * gate_w[e] since IN==1)
  float lg[8], m = -3.0e38f;
#pragma unroll
  for (int e = 0; e < 8; ++e) { lg[e] = x * gate[e]; m = fmaxf(m, lg[e]); }
  float s = 0.0f;
#pragma unroll
  for (int e = 0; e < 8; ++e) { lg[e] = __expf(lg[e] - m); s += lg[e]; }
  float inv_s = 1.0f / s;
#pragma unroll
  for (int o = 0; o < 32; ++o) out[o] = 0.0f;
#pragma unroll
  for (int e = 0; e < 8; ++e) {
    float ge = lg[e] * inv_s;
    float w  = ge * sil;
    float u[8];
#pragma unroll
    for (int c = 0; c < 8; ++c) u[c] = ge * bs[c];
    const float* bw = base + e * 32;
    const float* sw = spline + e * 32 * 8;
#pragma unroll
    for (int o = 0; o < 32; ++o) {
      float acc = w * bw[o];
      const float* swo = sw + o * 8;
#pragma unroll
      for (int c = 0; c < 8; ++c) acc += u[c] * swo[c];
      out[o] += acc;
    }
  }
}

// Weight permute (f32 [64][64][5] -> bf16 [64][320] K-major) + zero pad rows.
__global__ void prep_kernel(const float* __restrict__ conv_w,
                            unsigned short* __restrict__ wmat,
                            unsigned short* __restrict__ padded) {
  int idx = blockIdx.x * 256 + threadIdx.x;
  if (idx < 64 * KTOT) {
    int o = idx / KTOT, r = idx % KTOT;
    int k = r / 64, i = r % 64;
    wmat[idx] = f32_to_bf16(conv_w[(o * 64 + i) * 5 + k]);
  } else {
    int z = idx - 64 * KTOT;            // 32 batches * 4 pad rows * 64 ch = 8192
    int b = z / 256, rr = z % 256;
    int pr = rr / 64;
    int prow = (pr < 2) ? pr : (T_SZ + pr);
    padded[((size_t)b * TPAD + prow) * OUTC + (rr % 64)] = 0;
  }
}

// One thread per (b,t): evaluate both KAN mixtures, write bf16 comb row.
__global__ void __launch_bounds__(256)
kan_kernel(const float* __restrict__ a, const float* __restrict__ d,
           const float* __restrict__ gate_a, const float* __restrict__ base_a,
           const float* __restrict__ spline_a,
           const float* __restrict__ gate_d, const float* __restrict__ base_d,
           const float* __restrict__ spline_d,
           unsigned short* __restrict__ padded) {
  __shared__ float sg[2][8];
  __shared__ float sb[2][256];
  __shared__ float ss[2][2048];
  for (int i = threadIdx.x; i < 8; i += 256)    { sg[0][i] = gate_a[i];   sg[1][i] = gate_d[i]; }
  for (int i = threadIdx.x; i < 256; i += 256)  { sb[0][i] = base_a[i];   sb[1][i] = base_d[i]; }
  for (int i = threadIdx.x; i < 2048; i += 256) { ss[0][i] = spline_a[i]; ss[1][i] = spline_d[i]; }
  __syncthreads();

  int idx = blockIdx.x * 256 + threadIdx.x;
  int b = idx / T_SZ, t = idx % T_SZ;
  float xa = a[idx], xd = d[idx];
  unsigned short* row = padded + ((size_t)b * TPAD + (t + 2)) * OUTC;

  float out[32];
  kan_eval(xa, sg[0], sb[0], ss[0], out);
#pragma unroll
  for (int o = 0; o < 32; ++o) row[o] = f32_to_bf16(out[o]);
  kan_eval(xd, sg[1], sb[1], ss[1], out);
#pragma unroll
  for (int o = 0; o < 32; ++o) row[32 + o] = f32_to_bf16(out[o]);
}

// Implicit-GEMM conv1d(k=5, SAME) via v_wmma_f32_16x16x32_bf16.
// Block = 8 waves; wave w owns 16 output rows; N=64 -> 4 tiles; K=320 -> 10 steps.
__global__ void __launch_bounds__(256)
conv_kernel(const unsigned short* __restrict__ padded,
            const unsigned short* __restrict__ wmat,
            const float* __restrict__ bias,
            float* __restrict__ out) {
  __shared__ unsigned short lds_w[64 * KTOT];   // 40 KB of 320 KB WGP LDS
  {
    const uint4* src = (const uint4*)wmat;
    uint4* dst = (uint4*)lds_w;
#pragma unroll
    for (int i = threadIdx.x; i < (64 * KTOT * 2) / 16; i += 256) dst[i] = src[i];
  }
  __syncthreads();

  const int wave = threadIdx.x >> 5;
  const int lane = threadIdx.x & 31;
  const int l  = lane & 15;
  const int hi = lane >> 4;

  const int tiles_per_batch = T_SZ / 128;      // 64
  const int b  = blockIdx.x / tiles_per_batch;
  const int t0 = (blockIdx.x % tiles_per_batch) * 128 + wave * 16;

  // A row t = contiguous 320 bf16 window starting at padded row t (padded[t]=comb[t-2])
  const unsigned short* arow = padded + ((size_t)b * TPAD + (size_t)(t0 + l)) * OUTC;

  v8f acc[4] = {v8f{}, v8f{}, v8f{}, v8f{}};
#pragma unroll
  for (int ks = 0; ks < 10; ++ks) {
    const int k0 = ks * 32;
    // A, contiguous k order chosen here: lane group hi holds K 16*hi..16*hi+15; two 16B loads
    v8bf alo = *(const v8bf*)(arow + k0 + 16 * hi);
    v8bf ahi = *(const v8bf*)(arow + k0 + 16 * hi + 8);
    v16bf af = __builtin_shufflevector(alo, ahi, 0, 1, 2, 3, 4, 5, 6, 7,
                                       8, 9, 10, 11, 12, 13, 14, 15);
#pragma unroll
    for (int nt = 0; nt < 4; ++nt) {
      // B operand: N = lane col, K 16*hi..16*hi+15 in the same contiguous order as A; one 32B LDS load
      const v16bf bf = *(const v16bf*)(lds_w + (nt * 16 + l) * KTOT + k0 + 16 * hi);
      acc[nt] = __builtin_amdgcn_wmma_f32_16x16x32_bf16(
          false, af, false, bf, (short)0, acc[nt], false, false);
    }
  }

  // Epilogue: bias + ReLU, f32 out [B][T][64]. C/D layout: M = r + 8*hi, N = l.
#pragma unroll
  for (int nt = 0; nt < 4; ++nt) {
    const int o = nt * 16 + l;
    const float bv = bias[o];
#pragma unroll
    for (int r = 0; r < 8; ++r) {
      const int t = t0 + r + 8 * hi;
      float v = acc[nt][r] + bv;
      out[((size_t)b * T_SZ + t) * OUTC + o] = v > 0.0f ? v : 0.0f;
    }
  }
}

extern "C" void kernel_launch(void* const* d_in, const int* in_sizes, int n_in,
                              void* d_out, int out_size, void* d_ws, size_t ws_size,
                              hipStream_t stream) {
  const float* a        = (const float*)d_in[0];
  const float* d        = (const float*)d_in[1];
  const float* gate_a   = (const float*)d_in[2];
  const float* base_a   = (const float*)d_in[3];
  const float* spline_a = (const float*)d_in[4];
  const float* gate_d   = (const float*)d_in[5];
  const float* base_d   = (const float*)d_in[6];
  const float* spline_d = (const float*)d_in[7];
  const float* conv_w   = (const float*)d_in[8];
  const float* conv_b   = (const float*)d_in[9];

  // ws layout: bf16 padded comb [B][T+4][64], then bf16 wmat_t [64][320]
  const size_t padded_bytes = (size_t)B_SZ * TPAD * OUTC * sizeof(unsigned short);
  unsigned short* padded = (unsigned short*)d_ws;
  unsigned short* wmat   = (unsigned short*)((char*)d_ws + padded_bytes);

  prep_kernel<<<(64 * KTOT + 32 * 4 * 64) / 256, 256, 0, stream>>>(conv_w, wmat, padded);
  kan_kernel<<<(B_SZ * T_SZ) / 256, 256, 0, stream>>>(
      a, d, gate_a, base_a, spline_a, gate_d, base_d, spline_d, padded);
  conv_kernel<<<B_SZ * (T_SZ / 128), 256, 0, stream>>>(padded, wmat, conv_b, (float*)d_out);
}
